// GIN_27212912787986
// MI455X (gfx1250) — matrix operand order repaired and checked by reading, not compiled, not run on hardware
//
#include <hip/hip_runtime.h>
#include <hip/hip_bf16.h>

// ---------------------------------------------------------------------------
// GIN 2-layer forward for MI455X (gfx1250, wave32, WMMA).
//   agg = segment_sum(x[src], dst);  h = relu(((1+eps)x + agg) @ W1 + b1)
//   h   = dropout(h, 0.5) * 2
//   agg = segment_sum(h[src], dst);  out = ((1+eps)h + agg) @ W2 + b2
// N=50000, E=600000, F_IN=HID=128, OUT=64.
//
// Roofline: scatter phase dominates (~77M f32 L2 atomics per layer; x is
// 25.6MB and L2-resident at 192MB). GEMMs total 2.5 GFLOP and run on
// v_wmma_f32_16x16x32_bf16 (fp32 accumulate). Weights are transposed +
// down-converted to bf16 once so B fragments are contiguous 32B loads.
// ---------------------------------------------------------------------------

#define GIN_N   50000
#define GIN_E   600000
#define GIN_F   128      // F_IN == HID
#define GIN_OUT 64
#define GIN_EPS 1e-9f

typedef __attribute__((ext_vector_type(16))) __bf16 v16bf;
typedef __attribute__((ext_vector_type(8)))  float  v8f;

// Deterministic "dropout" mask (p=0.5). JAX threefry bits are not
// reproducible here; this is a fixed integer hash -> same output every call.
__device__ __forceinline__ bool gin_keep(unsigned idx) {
    unsigned x = idx * 2654435761u ^ 0x9E3779B9u;
    x ^= x >> 16; x *= 0x7feb352du;
    x ^= x >> 15; x *= 0x846ca68bu;
    x ^= x >> 16;
    return (x & 1u) != 0u;
}

// Wt[n*128 + k] = bf16(W[k*NC + n]); pre-transpose so WMMA B fragments are
// contiguous per lane.
__global__ void gin_wt_bf16(const float* __restrict__ W, __bf16* __restrict__ Wt,
                            int NC) {
    const int i = blockIdx.x * blockDim.x + threadIdx.x;
    if (i >= GIN_F * NC) return;
    const int n = i >> 7;            // / 128
    const int k = i & (GIN_F - 1);
    Wt[i] = (__bf16)W[(size_t)k * NC + n];
}

// One wave (32 lanes) per edge; each lane moves 4 consecutive floats.
// Gathers hit L2 (x is 25.6MB << 192MB L2); atomics resolve in L2.
__global__ void gin_scatter_add(const float* __restrict__ X,
                                const int*   __restrict__ ei,  // [2, E]
                                float*       __restrict__ AGG) {
    const int g = blockIdx.x * blockDim.x + threadIdx.x;
    const int e = g >> 5;
    if (e >= GIN_E) return;
    const int lane = g & 31;
    const int s = ei[e];           // src
    const int d = ei[GIN_E + e];   // dst
    const float4 v = *reinterpret_cast<const float4*>(X + (size_t)s * GIN_F + lane * 4);
    float* o = AGG + (size_t)d * GIN_F + lane * 4;
    atomicAdd(o + 0, v.x);
    atomicAdd(o + 1, v.y);
    atomicAdd(o + 2, v.z);
    atomicAdd(o + 3, v.w);
}

// Fused GIN linear: OUT[16 x NC] = ((1+eps)*X + AGG)[16 x 128] @ W[128 x NC] + bias
// One block per 16-row node tile; wave w owns output columns [16w, 16w+16).
// A-tile staged in LDS as bf16 (stride padded -> conflict-free ds_load_b128).
// B fragments come from pre-transposed bf16 W^T: 32B contiguous per lane.
// RELU_DROP=1 applies relu + deterministic 0.5-dropout (x2 scale).
template <int NC, int RELU_DROP>
__global__ void gin_mm_wmma(const float*  __restrict__ X,
                            const float*  __restrict__ AGG,
                            const __bf16* __restrict__ Wt,    // [NC x 128] bf16
                            const float*  __restrict__ BIAS,  // [NC]
                            float*        __restrict__ OUT) { // [N x NC]
    constexpr int LDSP = GIN_F + 8;          // 272B row stride -> conflict-free
    __shared__ __bf16 lds[16][LDSP];

    const int row0 = blockIdx.x * 16;
    const int tid  = threadIdx.x;
    const int nth  = blockDim.x;             // (NC/16) * 32

    // Stage xe = (1+eps)*x + agg as bf16.
    for (int i = tid; i < 16 * GIN_F; i += nth) {
        const int r = i >> 7;                // / 128
        const int c = i & (GIN_F - 1);
        const size_t off = (size_t)(row0 + r) * GIN_F + c;
        lds[r][c] = (__bf16)((1.0f + GIN_EPS) * X[off] + AGG[off]);
    }
    __syncthreads();

    const int wave = tid >> 5;
    const int lane = tid & 31;
    const int m    = lane & 15;              // A row within tile
    const int hh   = lane >> 4;              // half-wave select
    const int col  = wave * 16 + (lane & 15);

    // Lane's B stream: Wt[col][kk + 16*hh + e], e=0..15 contiguous, 32B aligned.
    const __bf16* bsrc = Wt + (size_t)col * GIN_F + 16 * hh;

    v8f acc = {};
    #pragma unroll
    for (int kk = 0; kk < GIN_F; kk += 32) { // 4 WMMA per tile
        v16bf a;
        // A 16x32 bf16 layout: elem e -> K = kk+16*hh+e (matches B)
        #pragma unroll
        for (int e = 0; e < 8; ++e) {
            a[e]     = lds[m][kk + 16 * hh + e];
            a[e + 8] = lds[m][kk + 16 * hh + 8 + e];
        }
        // B 32x16 bf16 layout: lane column = lane&15, elem e -> K = kk+16*hh+e
        const v16bf b = *reinterpret_cast<const v16bf*>(bsrc + kk);
        acc = __builtin_amdgcn_wmma_f32_16x16x32_bf16(
            /*neg_a=*/false, a, /*neg_b=*/false, b,
            /*c_mod=*/(short)0, acc, /*reuse_a=*/false, /*reuse_b=*/false);
    }

    const float bias = BIAS[col];
    // C/D layout: VGPR r -> row (r + 8*hh), col = lane&15.
    #pragma unroll
    for (int r = 0; r < 8; ++r) {
        const int row = row0 + r + 8 * hh;
        float v = acc[r] + bias;
        if (RELU_DROP) {
            v = v > 0.0f ? v : 0.0f;
            const unsigned idx = (unsigned)row * (unsigned)NC + (unsigned)col;
            v = gin_keep(idx) ? v * 2.0f : 0.0f;  // / (1 - 0.5)
        }
        OUT[(size_t)row * NC + col] = v;
    }
}

extern "C" void kernel_launch(void* const* d_in, const int* in_sizes, int n_in,
                              void* d_out, int out_size, void* d_ws, size_t ws_size,
                              hipStream_t stream) {
    const float* x  = (const float*)d_in[0];   // [N, 128]
    const int*   ei = (const int*)  d_in[1];   // [2, E]
    const float* W1 = (const float*)d_in[2];   // [128, 128]
    const float* b1 = (const float*)d_in[3];   // [128]
    const float* W2 = (const float*)d_in[4];   // [128, 64]
    const float* b2 = (const float*)d_in[5];   // [64]
    float* out = (float*)d_out;                // [N, 64]

    // Workspace layout: agg | h | W1t(bf16) | W2t(bf16)  (~51.25 MB total)
    float*  agg = (float*)d_ws;                         // N*128 f32
    float*  h   = agg + (size_t)GIN_N * GIN_F;          // N*128 f32
    __bf16* w1t = (__bf16*)(h + (size_t)GIN_N * GIN_F); // 128*128 bf16
    __bf16* w2t = w1t + GIN_F * GIN_F;                  // 128*64  bf16
    const size_t aggBytes = (size_t)GIN_N * GIN_F * sizeof(float);

    const int scatterBlocks = (GIN_E * 32 + 255) / 256;
    const int rowTiles      = GIN_N / 16;               // 3125

    // ---- Weight prep (once per launch; tiny) ----
    gin_wt_bf16<<<(GIN_F * GIN_F + 255) / 256, 256, 0, stream>>>(W1, w1t, GIN_F);
    gin_wt_bf16<<<(GIN_F * GIN_OUT + 255) / 256, 256, 0, stream>>>(W2, w2t, GIN_OUT);

    // ---- Layer 1 ----
    hipMemsetAsync(agg, 0, aggBytes, stream);
    gin_scatter_add<<<scatterBlocks, 256, 0, stream>>>(x, ei, agg);
    gin_mm_wmma<GIN_F, 1><<<rowTiles, (GIN_F / 16) * 32, 0, stream>>>(
        x, agg, w1t, b1, h);

    // ---- Layer 2 (reuse agg buffer) ----
    hipMemsetAsync(agg, 0, aggBytes, stream);
    gin_scatter_add<<<scatterBlocks, 256, 0, stream>>>(h, ei, agg);
    gin_mm_wmma<GIN_OUT, 0><<<rowTiles, (GIN_OUT / 16) * 32, 0, stream>>>(
        h, agg, w2t, b2, out);
}
